// FFMCell_56435870269664
// MI455X (gfx1250) — matrix operand order repaired and checked by reading, not compiled, not run on hardware
//
#include <hip/hip_runtime.h>
#include <hip/hip_bf16.h>

// ---------------- problem constants ----------------
#define T_      1024
#define B_      32
#define D_      256
#define TRACE_  64
#define CTX_    16
#define OUT_    256
#define M_      (T_ * B_)           // 32768 rows
#define KZ_     (2 * TRACE_ * CTX_) // 2048

// ---------------- WMMA vector types ----------------
typedef __attribute__((ext_vector_type(16))) __bf16 v16bf;
typedef __attribute__((ext_vector_type(8)))  float  v8f;
typedef unsigned int u32x4 __attribute__((ext_vector_type(4)));
typedef int          i32x4 __attribute__((ext_vector_type(4)));
typedef int          i32x8 __attribute__((ext_vector_type(8)));

union V16U { v16bf v; uint4 q[2]; };

#if __has_builtin(__builtin_amdgcn_tensor_load_to_lds)
#define USE_TDM 1
#else
#define USE_TDM 0
#endif

// float -> bf16 (round to nearest even)
static __device__ __forceinline__ unsigned short f2bf(float f) {
    unsigned int u = __float_as_uint(f);
    unsigned int r = 0x7FFFu + ((u >> 16) & 1u);
    return (unsigned short)((u + r) >> 16);
}

static __device__ __forceinline__ float sigmf(float x) {
    return 1.0f / (1.0f + __expf(-x));
}

// A fragment (16x32 bf16) from row-major memory, row stride ldk elements.
// Lane L: row m; K = kb..kb+7 and kb+16..kb+23, kb=(L>>4)*8 -> two 16B loads.
static __device__ __forceinline__ v16bf load_a_frag(const unsigned short* base, int ldk,
                                                    int m, int k0, int lane) {
    const unsigned short* p = base + (size_t)m * ldk + k0 + ((lane >> 4) * 8);
    V16U u;
    u.q[0] = *(const uint4*)p;
    u.q[1] = *(const uint4*)(p + 16);
    return u.v;
}

// A fragment from an LDS-staged [rows][256] bf16 tile.
static __device__ __forceinline__ v16bf lds_a_frag(const unsigned short* sb,
                                                   int row, int k0, int lane) {
    const unsigned short* p = sb + row * 256 + k0 + ((lane >> 4) * 8);
    V16U u;
    u.q[0] = *(const uint4*)p;
    u.q[1] = *(const uint4*)(p + 16);
    return u.v;
}

// B fragment (32x16 bf16) from N-major (transposed) weights wt[N][K].
// Lane L: column n0+(L&15); K = kb..kb+7 and kb+16..kb+23, kb=k0+(L>>4)*8 -> two 16B loads.
static __device__ __forceinline__ v16bf load_b_frag(const unsigned short* wt, int ldk,
                                                    int n0, int k0, int lane) {
    const unsigned short* p = wt + (size_t)(n0 + (lane & 15)) * ldk + k0 + ((lane >> 4) * 8);
    V16U u;
    u.q[0] = *(const uint4*)p;
    u.q[1] = *(const uint4*)(p + 16);
    return u.v;
}

#define WMMA_BF16(A, Bv, C) \
    __builtin_amdgcn_wmma_f32_16x16x32_bf16(false, (A), false, (Bv), (short)0, (C), false, false)

#if USE_TDM
// Tensor Data Mover: async DMA of a 2-D bf16 tile (tile_k x tile_rows) from a
// row-major tensor (row length tensor_k, stride stride_k, tensor_rows rows)
// into LDS at byte offset lds_off (rows stored contiguously, tile_k*2 B each).
// D# built per CDNA5 ISA (group0/group1, data_size=2B, type=2).
static __device__ __forceinline__ void tdm_load_2d(unsigned lds_off, const void* gp,
                                                   unsigned tile_k, unsigned tile_rows,
                                                   unsigned tensor_k, unsigned tensor_rows,
                                                   unsigned stride_k) {
    unsigned long long ga = (unsigned long long)gp;
    u32x4 g0;
    g0.x = 1u;                                   // count=1, user descriptor
    g0.y = lds_off;                              // LDS byte address
    g0.z = (unsigned)ga;                         // global_addr[31:0]
    g0.w = (unsigned)((ga >> 32) & 0x01FFFFFFull) | 0x80000000u; // addr[56:32] | type=2
    i32x8 g1;
    g1[0] = 0x00010000;                          // data_size=1 (2 bytes), no flags
    g1[1] = (int)((tensor_k & 0xFFFFu) << 16);   // tensor_dim0[15:0]
    g1[2] = (int)((tensor_k >> 16) | ((tensor_rows & 0xFFFFu) << 16));
    g1[3] = (int)((tensor_rows >> 16) | (tile_k << 16));  // tile_dim0
    g1[4] = (int)(tile_rows & 0xFFFFu);          // tile_dim1 (tile_dim2=0)
    g1[5] = (int)stride_k;                       // tensor_dim0_stride[31:0]
    g1[6] = 0;                                   // stride[47:32] | dim1_stride lo
    g1[7] = 0;
    i32x4 gz = {0, 0, 0, 0};
#if __has_include(<hip/amd_detail/amd_gfx1250_TDM.h>)
    i32x8 gz8 = {};
    __builtin_amdgcn_tensor_load_to_lds(g0, g1, gz, gz, gz8, 0);
#else
    __builtin_amdgcn_tensor_load_to_lds(g0, g1, gz, gz, 0);
#endif
}
#endif

// ---------------- kernel 0a: f32 -> bf16 pack ----------------
__global__ void k_pack_x(const float* __restrict__ x, unsigned short* __restrict__ xb, int n) {
    int i = blockIdx.x * blockDim.x + threadIdx.x;
    if (i < n) xb[i] = f2bf(x[i]);
}

// ---------------- kernel 0b: weight transpose K x N -> N x K, bf16 ----------------
__global__ void k_trans_w(const float* __restrict__ w, unsigned short* __restrict__ wt,
                          int K, int N) {
    int i = blockIdx.x * blockDim.x + threadIdx.x;
    if (i >= K * N) return;
    int n = i % N;
    int k = i / N;
    wt[(size_t)n * K + k] = f2bf(w[(size_t)k * N + n]);
}

// ---------------- kernel 1: fused input GEMMs ----------------
// gated[m, tr] = (x@w_pre + b_pre) * sigmoid(x@w_gate_in + b_gate_in)
__global__ void k_gemm_in(const unsigned short* __restrict__ xb,
                          const unsigned short* __restrict__ wpre_t,
                          const unsigned short* __restrict__ wgi_t,
                          const float* __restrict__ b_pre,
                          const float* __restrict__ b_gi,
                          float* __restrict__ gated) {
    const int lane = threadIdx.x & 31;
    const int wave = threadIdx.x >> 5;
    const int m0 = blockIdx.x * 128 + wave * 16;
    const int mrow = m0 + (lane & 15);

    v8f accP[4], accG[4];
#pragma unroll
    for (int j = 0; j < 4; ++j) { accP[j] = {}; accG[j] = {}; }

    for (int k0 = 0; k0 < D_; k0 += 32) {
        v16bf a = load_a_frag(xb, D_, mrow, k0, lane);
#pragma unroll
        for (int j = 0; j < 4; ++j) {
            v16bf bp = load_b_frag(wpre_t, D_, j * 16, k0, lane);
            accP[j] = WMMA_BF16(a, bp, accP[j]);
            v16bf bg = load_b_frag(wgi_t, D_, j * 16, k0, lane);
            accG[j] = WMMA_BF16(a, bg, accG[j]);
        }
    }

    const int nl = lane & 15;
    const int mh = (lane >> 4) * 8;
#pragma unroll
    for (int j = 0; j < 4; ++j) {
        float bpv = b_pre[j * 16 + nl];
        float bgv = b_gi[j * 16 + nl];
#pragma unroll
        for (int r = 0; r < 8; ++r) {
            int m = m0 + mh + r;
            gated[(size_t)m * TRACE_ + j * 16 + nl] =
                (accP[j][r] + bpv) * sigmf(accG[j][r] + bgv);
        }
    }
}

// ---------------- kernel 2: sequential complex scan over T ----------------
__global__ void k_scan(const float* __restrict__ gated,
                       const int* __restrict__ start,
                       const float* __restrict__ st_re,
                       const float* __restrict__ st_im,
                       const float* __restrict__ ffm_a,
                       const float* __restrict__ ffm_b,
                       unsigned short* __restrict__ z_in,
                       float* __restrict__ fstate) {
    const int tid = blockIdx.x * blockDim.x + threadIdx.x;   // < 32768
    const int b  = tid >> 10;
    const int rm = tid & 1023;
    const int tr = rm >> 4;
    const int cx = rm & 15;

    const float d  = __expf(-fabsf(ffm_a[tr]));
    const float bb = ffm_b[cx];
    const float cr = d * __cosf(bb);
    const float ci = d * __sinf(bb);

    float sre = st_re[tid];
    float sim = st_im[tid];

    for (int t = 0; t < T_; ++t) {
        const int row = t * B_ + b;
        if (start[row]) { sre = 0.0f; sim = 0.0f; }
        const float g = gated[(size_t)row * TRACE_ + tr];
        const float nr = fmaf(sre, cr, fmaf(-sim, ci, g));
        const float ni = fmaf(sre, ci, sim * cr);
        sre = nr; sim = ni;
        const size_t base = (size_t)row * KZ_ + tr * 32 + cx;
        z_in[base]      = f2bf(sre);
        z_in[base + 16] = f2bf(sim);
    }
    fstate[2 * tid]     = sre;    // final_state, complex64 interleaved
    fstate[2 * tid + 1] = sim;
}

// ---------------- kernel 3: fused output GEMMs + layernorm + residual ----------------
// 32 rows/block, 8 waves, each wave: 2 M-tiles x 2 N-tiles.
// A tiles (z_in chunks + x tile) staged into LDS by TDM, double-buffered.
__global__ void k_gemm_out(const unsigned short* __restrict__ z_in,
                           const unsigned short* __restrict__ xb,
                           const unsigned short* __restrict__ wz_t,
                           const unsigned short* __restrict__ wgo_t,
                           const unsigned short* __restrict__ wsk_t,
                           const float* __restrict__ b_z,
                           const float* __restrict__ b_go,
                           const float* __restrict__ b_sk,
                           float* __restrict__ out) {
    extern __shared__ __align__(16) char smem[];            // 96 KB dynamic
    unsigned short* abuf = (unsigned short*)smem;            // [2][32][256] bf16, 32 KB
    unsigned short* xbuf = (unsigned short*)(smem + 32768);  // [32][256] bf16, 16 KB
    float* lz = (float*)smem;                                // epilogue aliases, 3x32 KB
    float* lg = (float*)(smem + 32768);
    float* ls = (float*)(smem + 65536);

    const int lane = threadIdx.x & 31;
    const int wave = threadIdx.x >> 5;
    const int m0 = blockIdx.x * 32;
    const int n0 = wave * 32;
    const int n1 = n0 + 16;
    const int rlo = lane & 15;

    v8f Zc[2][2], Gc[2][2], Sc[2][2];
#pragma unroll
    for (int i = 0; i < 2; ++i)
#pragma unroll
        for (int j = 0; j < 2; ++j) { Zc[i][j] = {}; Gc[i][j] = {}; Sc[i][j] = {}; }

#if USE_TDM
    if (threadIdx.x < 32) {   // one wave drives the DMA engine
        tdm_load_2d((unsigned)(size_t)(void*)xbuf, xb + (size_t)m0 * D_,
                    256, 32, D_, M_, D_);
        tdm_load_2d((unsigned)(size_t)(void*)abuf, z_in + (size_t)m0 * KZ_,
                    256, 32, KZ_, M_, KZ_);
    }
#endif

    for (int c = 0; c < 8; ++c) {
#if USE_TDM
        if (threadIdx.x < 32) {
            if (c < 7) {
                tdm_load_2d((unsigned)(size_t)(void*)(abuf + ((c + 1) & 1) * 8192),
                            z_in + ((size_t)m0 * KZ_ + (c + 1) * 256),
                            256, 32, KZ_, M_, KZ_);
                __builtin_amdgcn_s_wait_tensorcnt((short)1);  // chunk c (and x) done
            } else {
                __builtin_amdgcn_s_wait_tensorcnt((short)0);
            }
        }
        __syncthreads();
#else
        // cooperative fallback staging (single-buffered)
        __syncthreads();
        {
            const unsigned short* src = z_in + (size_t)m0 * KZ_ + c * 256;
            unsigned short* dst = abuf;
#pragma unroll
            for (int v = 0; v < 4; ++v) {
                int u = threadIdx.x * 4 + v;       // 1024 uint4 = 32 rows x 32 uint4
                int row = u >> 5, col8 = (u & 31) * 8;
                *(uint4*)(dst + row * 256 + col8) =
                    *(const uint4*)(src + (size_t)row * KZ_ + col8);
                if (c == 0)
                    *(uint4*)(xbuf + row * 256 + col8) =
                        *(const uint4*)(xb + (size_t)(m0 + row) * D_ + col8);
            }
        }
        __syncthreads();
#endif
#if USE_TDM
        const unsigned short* ab = abuf + (c & 1) * 8192;
#else
        const unsigned short* ab = abuf;
#endif
        for (int kk = 0; kk < 256; kk += 32) {
            v16bf a0 = lds_a_frag(ab, rlo,      kk, lane);
            v16bf a1 = lds_a_frag(ab, 16 + rlo, kk, lane);
            const int kg = c * 256 + kk;
            v16bf bz0 = load_b_frag(wz_t, KZ_, n0, kg, lane);
            v16bf bz1 = load_b_frag(wz_t, KZ_, n1, kg, lane);
            Zc[0][0] = WMMA_BF16(a0, bz0, Zc[0][0]);
            Zc[0][1] = WMMA_BF16(a0, bz1, Zc[0][1]);
            Zc[1][0] = WMMA_BF16(a1, bz0, Zc[1][0]);
            Zc[1][1] = WMMA_BF16(a1, bz1, Zc[1][1]);
            if (c == 0) {   // fold K=256 gate/skip GEMMs into first chunk
                v16bf ax0 = lds_a_frag(xbuf, rlo,      kk, lane);
                v16bf ax1 = lds_a_frag(xbuf, 16 + rlo, kk, lane);
                v16bf bg0 = load_b_frag(wgo_t, D_, n0, kk, lane);
                v16bf bg1 = load_b_frag(wgo_t, D_, n1, kk, lane);
                Gc[0][0] = WMMA_BF16(ax0, bg0, Gc[0][0]);
                Gc[0][1] = WMMA_BF16(ax0, bg1, Gc[0][1]);
                Gc[1][0] = WMMA_BF16(ax1, bg0, Gc[1][0]);
                Gc[1][1] = WMMA_BF16(ax1, bg1, Gc[1][1]);
                v16bf bs0 = load_b_frag(wsk_t, D_, n0, kk, lane);
                v16bf bs1 = load_b_frag(wsk_t, D_, n1, kk, lane);
                Sc[0][0] = WMMA_BF16(ax0, bs0, Sc[0][0]);
                Sc[0][1] = WMMA_BF16(ax0, bs1, Sc[0][1]);
                Sc[1][0] = WMMA_BF16(ax1, bs0, Sc[1][0]);
                Sc[1][1] = WMMA_BF16(ax1, bs1, Sc[1][1]);
            }
        }
        __syncthreads();   // done reading abuf[(c&1)] before it is overwritten
    }

    // epilogue: write accumulators (+bias, sigmoid) to LDS (aliases DMA buffers)
    const int nl = lane & 15;
    const int mh = (lane >> 4) * 8;
    const float bzv[2] = { b_z[n0 + nl],  b_z[n1 + nl]  };
    const float bgv[2] = { b_go[n0 + nl], b_go[n1 + nl] };
    const float bsv[2] = { b_sk[n0 + nl], b_sk[n1 + nl] };
#pragma unroll
    for (int mt = 0; mt < 2; ++mt) {
#pragma unroll
        for (int nt = 0; nt < 2; ++nt) {
            const int nc = (nt ? n1 : n0) + nl;
#pragma unroll
            for (int r = 0; r < 8; ++r) {
                const int ml = mt * 16 + mh + r;
                lz[ml * 256 + nc] = Zc[mt][nt][r] + bzv[nt];
                lg[ml * 256 + nc] = sigmf(Gc[mt][nt][r] + bgv[nt]);
                ls[ml * 256 + nc] = Sc[mt][nt][r] + bsv[nt];
            }
        }
    }
    __syncthreads();

    // per-row layernorm: 32 rows, 8 threads/row, 32 cols each
    const int row = threadIdx.x >> 3;
    const int seg = threadIdx.x & 7;
    float sum = 0.0f, ssq = 0.0f;
#pragma unroll
    for (int i = 0; i < 32; ++i) {
        const int cc = seg * 32 + i;
        const float y = lz[row * 256 + cc] * lg[row * 256 + cc];
        sum += y;
        ssq += y * y;
    }
#pragma unroll
    for (int msk = 4; msk >= 1; msk >>= 1) {
        sum += __shfl_xor(sum, msk, 8);
        ssq += __shfl_xor(ssq, msk, 8);
    }
    const float mu   = sum * (1.0f / 256.0f);
    const float var  = ssq * (1.0f / 256.0f) - mu * mu;
    const float rinv = rsqrtf(var + 1e-6f);

    const size_t ob = (size_t)(m0 + row) * 256;
#pragma unroll
    for (int i = 0; i < 32; ++i) {
        const int cc = seg * 32 + i;
        const float g = lg[row * 256 + cc];
        const float y = lz[row * 256 + cc] * g;
        out[ob + cc] = (y - mu) * rinv + ls[row * 256 + cc] * (1.0f - g);
    }
}

// ---------------- host launcher ----------------
extern "C" void kernel_launch(void* const* d_in, const int* in_sizes, int n_in,
                              void* d_out, int out_size, void* d_ws, size_t ws_size,
                              hipStream_t stream) {
    const float* x         = (const float*)d_in[0];
    const int*   start     = (const int*)  d_in[1];
    const float* state_re  = (const float*)d_in[2];
    const float* state_im  = (const float*)d_in[3];
    const float* w_gate_in = (const float*)d_in[4];
    const float* b_gate_in = (const float*)d_in[5];
    const float* w_pre     = (const float*)d_in[6];
    const float* b_pre     = (const float*)d_in[7];
    const float* w_z       = (const float*)d_in[8];
    const float* b_z       = (const float*)d_in[9];
    const float* w_gate_out= (const float*)d_in[10];
    const float* b_gate_out= (const float*)d_in[11];
    const float* w_skip    = (const float*)d_in[12];
    const float* b_skip    = (const float*)d_in[13];
    const float* ffm_a     = (const float*)d_in[14];
    const float* ffm_b     = (const float*)d_in[15];

    float* fstate = (float*)d_out;                        // 2*B*TRACE*CTX floats
    float* out    = (float*)d_out + 2 * B_ * TRACE_ * CTX_;

    char* ws = (char*)d_ws;
    size_t off = 0;
    auto carve = [&](size_t bytes) -> void* {
        void* p = ws + off;
        off += (bytes + 255) & ~(size_t)255;
        return p;
    };
    unsigned short* x_bf   = (unsigned short*)carve((size_t)M_ * D_ * 2);
    unsigned short* wgi_t  = (unsigned short*)carve((size_t)TRACE_ * D_ * 2);
    unsigned short* wpre_t = (unsigned short*)carve((size_t)TRACE_ * D_ * 2);
    unsigned short* wgo_t  = (unsigned short*)carve((size_t)OUT_ * D_ * 2);
    unsigned short* wsk_t  = (unsigned short*)carve((size_t)OUT_ * D_ * 2);
    unsigned short* wz_t   = (unsigned short*)carve((size_t)OUT_ * KZ_ * 2);
    float*          gated  = (float*)carve((size_t)M_ * TRACE_ * 4);
    unsigned short* z_in   = (unsigned short*)carve((size_t)M_ * KZ_ * 2);
    (void)ws_size;

    {
        int n = M_ * D_;
        k_pack_x<<<(n + 255) / 256, 256, 0, stream>>>(x, x_bf, n);
    }
    k_trans_w<<<(D_ * TRACE_ + 255) / 256, 256, 0, stream>>>(w_gate_in, wgi_t, D_, TRACE_);
    k_trans_w<<<(D_ * TRACE_ + 255) / 256, 256, 0, stream>>>(w_pre,     wpre_t, D_, TRACE_);
    k_trans_w<<<(D_ * OUT_   + 255) / 256, 256, 0, stream>>>(w_gate_out, wgo_t, D_, OUT_);
    k_trans_w<<<(D_ * OUT_   + 255) / 256, 256, 0, stream>>>(w_skip,     wsk_t, D_, OUT_);
    k_trans_w<<<(KZ_ * OUT_  + 255) / 256, 256, 0, stream>>>(w_z,        wz_t, KZ_, OUT_);

    k_gemm_in<<<M_ / 128, 256, 0, stream>>>(x_bf, wpre_t, wgi_t, b_pre, b_gate_in, gated);

    k_scan<<<(B_ * TRACE_ * CTX_) / 256, 256, 0, stream>>>(
        gated, start, state_re, state_im, ffm_a, ffm_b, z_in, fstate);

    k_gemm_out<<<M_ / 32, 256, 98304, stream>>>(
        z_in, x_bf, wz_t, wgo_t, wsk_t, b_z, b_gate_out, b_skip, out);

    (void)in_sizes; (void)n_in; (void)out_size;
}
